// SphericalSweep_28535762715157
// MI455X (gfx1250) — matrix operand order repaired and checked by reading, not compiled, not run on hardware
//
#include <hip/hip_runtime.h>
#include <cstdint>

// ---------- types ----------
typedef __attribute__((ext_vector_type(16))) __bf16 v16bf;
typedef __attribute__((ext_vector_type(8)))  float  v8f;

struct alignas(16) Q4 { uint32_t x0, x1, x2, x3; };
struct alignas(32) Q8 { Q4 lo, hi; };

// ---------- problem constants ----------
#define XD   32
#define HO   128
#define WOUT 256
#define NPTS (XD * HO * WOUT)            // 1,048,576 output spatial points
#define FC   64                          // CAM*C channels
#define WB_ELEMS (27 * 2 * 4 * 32 * 16)  // 110,592 bf16 weights, WMMA-A layout
#define VOL_BYTES ((size_t)NPTS * FC * 2) // 128 MB bf16 cost volume (L2-resident)
#define WBUF_OFF  VOL_BYTES
#define STATS_OFF (WBUF_OFF + (size_t)WB_ELEMS * 2)
#define ZPAGE_OFF (STATS_OFF + 512)      // 128B zero page for OOB B-tile lanes

static __device__ __forceinline__ uint16_t f2bf(float f) {
  uint32_t u = __float_as_uint(f);
  u += 0x7FFFu + ((u >> 16) & 1u);       // round-to-nearest-even
  return (uint16_t)(u >> 16);
}

// ---------- kernel 0: zero per-channel stats + zero page ----------
__global__ void zero_stats_k(float* st) {
  if (threadIdx.x < 160) st[threadIdx.x] = 0.0f;   // 128 stats + 32 floats zero page
}

// ---------- kernel 1: repack conv weights fp32 -> bf16 in WMMA A-matrix lane layout ----------
// flat idx = (((tap*2+kg)*4+mt)*32 + lane)*16 + i
// 16-bit A 16x32 layout (contiguous): lane<16: K = 0..15; lane>=16: K = 16..31
__global__ void repack_w_k(const float* __restrict__ w, uint16_t* __restrict__ wbuf) {
  int idx = blockIdx.x * blockDim.x + threadIdx.x;
  if (idx >= WB_ELEMS) return;
  int i    = idx & 15;
  int lane = (idx >> 4) & 31;
  int mt   = (idx >> 9) & 3;
  int kg   = (idx >> 11) & 1;
  int tap  = idx >> 12;
  int j = i >> 1, e = i & 1;
  int kbase = (j < 4) ? (2 * j) : (16 + 2 * (j - 4));
  int K = i + ((lane & 16) ? 16 : 0);
  int cin  = kg * 32 + K;
  int cout = mt * 16 + (lane & 15);
  wbuf[idx] = f2bf(w[((size_t)cout * 64 + cin) * 27 + tap]);
}

// ---------- kernel 2: multi-plane bilinear grid sample -> bf16 volume [pt][c0..c63] ----------
__global__ void sweep_sample_k(const float* __restrict__ feats,
                               const float* __restrict__ grids,
                               uint16_t* __restrict__ vol) {
  int tid = blockIdx.x * blockDim.x + threadIdx.x;
  int cam = tid & 3;
  int p   = tid >> 2;                    // p = x*HO*WOUT + ho*WOUT + wo
  const float* g = grids + (((size_t)cam * NPTS + p) << 1);
  float ix = (g[0] + 1.0f) * 64.0f - 0.5f;   // ((gx+1)*W - 1)*0.5, W=128
  float iy = (g[1] + 1.0f) * 64.0f - 0.5f;
  float x0f = floorf(ix), y0f = floorf(iy);
  int   x0 = (int)x0f, y0 = (int)y0f;
  float wx1 = ix - x0f, wx0 = 1.0f - wx1;
  float wy1 = iy - y0f, wy0 = 1.0f - wy1;
  const float* f = feats + (size_t)cam * 16 * 16384;

  float acc[16];
  #pragma unroll
  for (int c = 0; c < 16; ++c) acc[c] = 0.0f;

  int   xs[2]  = {x0, x0 + 1}, ys[2] = {y0, y0 + 1};
  float wxs[2] = {wx0, wx1},   wys[2] = {wy0, wy1};
  #pragma unroll
  for (int cy = 0; cy < 2; ++cy)
    #pragma unroll
    for (int cx = 0; cx < 2; ++cx) {
      int xi = xs[cx], yi = ys[cy];
      if (xi >= 0 && xi < 128 && yi >= 0 && yi < 128) {
        float wgt = wxs[cx] * wys[cy];
        int fi = yi * 128 + xi;
        #pragma unroll
        for (int c = 0; c < 16; ++c) acc[c] += wgt * f[c * 16384 + fi];
      }
    }

  Q4 lo, hi;
  lo.x0 = (uint32_t)f2bf(acc[0])  | ((uint32_t)f2bf(acc[1])  << 16);
  lo.x1 = (uint32_t)f2bf(acc[2])  | ((uint32_t)f2bf(acc[3])  << 16);
  lo.x2 = (uint32_t)f2bf(acc[4])  | ((uint32_t)f2bf(acc[5])  << 16);
  lo.x3 = (uint32_t)f2bf(acc[6])  | ((uint32_t)f2bf(acc[7])  << 16);
  hi.x0 = (uint32_t)f2bf(acc[8])  | ((uint32_t)f2bf(acc[9])  << 16);
  hi.x1 = (uint32_t)f2bf(acc[10]) | ((uint32_t)f2bf(acc[11]) << 16);
  hi.x2 = (uint32_t)f2bf(acc[12]) | ((uint32_t)f2bf(acc[13]) << 16);
  hi.x3 = (uint32_t)f2bf(acc[14]) | ((uint32_t)f2bf(acc[15]) << 16);
  Q4* dst = (Q4*)((char*)vol + (size_t)p * 128 + cam * 32);
  dst[0] = lo;
  dst[1] = hi;
}

// ---------- kernel 3: implicit-GEMM 3D conv via V_WMMA_F32_16X16X32_BF16 ----------
// block = 512 thr = 16 waves (4/SIMD32); wave computes 16 Wo points x 64 out channels.
// Weights (216 KB) async-copied to LDS once per block; B tiles read from L2-resident volume.
// OOB lanes load from a zero page (address select, no EXEC divergence).
__global__ void __launch_bounds__(512) conv3d_wmma_k(
    const uint16_t* __restrict__ volc,
    const uint16_t* __restrict__ wbuf,
    const float*    __restrict__ bias,
    float*          __restrict__ out,
    float*          __restrict__ gstats,
    const char*     __restrict__ zpage) {
  __shared__ __align__(32) unsigned char smem[WB_ELEMS * 2 + 512];
  uint16_t* wlds = (uint16_t*)smem;
  float*    csum = (float*)(smem + WB_ELEMS * 2);
  float*    csq  = csum + 64;

  int tid  = threadIdx.x;
  int lane = tid & 31;
  int wave = tid >> 5;                   // 0..15 -> 16-wide Wo strip
  int bid  = blockIdx.x;
  int ho   = bid & 127;
  int x    = bid >> 7;
  int wo_base = wave * 16;

  if (tid < 64) { csum[tid] = 0.0f; csq[tid] = 0.0f; }

  // async copy full weight buffer global -> LDS (13824 x b128); wlds sits at LDS offset 0
  {
    uint64_t src = (uint64_t)(const void*)wbuf;
    for (int it = tid; it < WB_ELEMS / 8; it += 512) {
      uint32_t lds_off = (uint32_t)(it * 16);
      uint32_t voff    = (uint32_t)(it * 16);
      asm volatile("global_load_async_to_lds_b128 %0, %1, %2"
                   :: "v"(lds_off), "v"(voff), "s"(src) : "memory");
    }
    asm volatile("s_wait_asynccnt 0" ::: "memory");
  }
  __syncthreads();

  const v16bf* wl = (const v16bf*)wlds;
  int nn = lane & 15;
  int hh = lane >> 4;                    // 0: K 0..15, 1: K 16..31 (B lane halves)
  int wvbase = wo_base + nn;
  uint32_t laneoff = (uint32_t)wvbase * 128u + (uint32_t)hh * 32u;
  const char* volb = (const char*)volc;
  const char* zp   = zpage + (uint32_t)hh * 32u;   // stays within 128B zero page

  v8f acc0 = {}, acc1 = {}, acc2 = {}, acc3 = {};

  int tile = lane;                       // running v16bf index: tap*256 + lane
  #pragma unroll 1
  for (int dz = -1; dz <= 1; ++dz) {
    int z = x + dz;
    #pragma unroll 1
    for (int dy = -1; dy <= 1; ++dy) {
      int y = ho + dy;
      bool zyok = ((unsigned)z < (unsigned)XD) & ((unsigned)y < (unsigned)HO);
      const char* base = volb + ((size_t)(uint32_t)(z * HO + y) << 15); // *WOUT*128B
      #pragma unroll 1
      for (int dxx = -1; dxx <= 1; ++dxx) {
        int wv = wvbase + dxx;
        bool ok = zyok & ((unsigned)wv < (unsigned)WOUT);
        // ---- B tiles (kg0, kg1): 4 unconditional global_load_b128; OOB -> zero page ----
        const char* p = ok ? (base + laneoff + dxx * 128) : zp;
        Q8 t0 = *(const Q8*)p;
        Q8 t1 = *(const Q8*)(p + 64);
        // ---- A tiles: 16 ds_load_b128, batched ahead of the WMMAs ----
        v16bf a00 = wl[tile +   0];
        v16bf a01 = wl[tile +  32];
        v16bf a02 = wl[tile +  64];
        v16bf a03 = wl[tile +  96];
        v16bf a10 = wl[tile + 128];
        v16bf a11 = wl[tile + 160];
        v16bf a12 = wl[tile + 192];
        v16bf a13 = wl[tile + 224];
        v16bf b0 = __builtin_bit_cast(v16bf, t0);
        v16bf b1 = __builtin_bit_cast(v16bf, t1);
        acc0 = __builtin_amdgcn_wmma_f32_16x16x32_bf16(false, a00, false, b0, (short)0, acc0, false, false);
        acc1 = __builtin_amdgcn_wmma_f32_16x16x32_bf16(false, a01, false, b0, (short)0, acc1, false, false);
        acc2 = __builtin_amdgcn_wmma_f32_16x16x32_bf16(false, a02, false, b0, (short)0, acc2, false, false);
        acc3 = __builtin_amdgcn_wmma_f32_16x16x32_bf16(false, a03, false, b0, (short)0, acc3, false, false);
        acc0 = __builtin_amdgcn_wmma_f32_16x16x32_bf16(false, a10, false, b1, (short)0, acc0, false, false);
        acc1 = __builtin_amdgcn_wmma_f32_16x16x32_bf16(false, a11, false, b1, (short)0, acc1, false, false);
        acc2 = __builtin_amdgcn_wmma_f32_16x16x32_bf16(false, a12, false, b1, (short)0, acc2, false, false);
        acc3 = __builtin_amdgcn_wmma_f32_16x16x32_bf16(false, a13, false, b1, (short)0, acc3, false, false);
        tile += 256;                     // next tap (256 v16bf tiles of 32 lanes)
        // pipeline shape: B loads -> A loads -> WMMAs (incremental dscnt waits)
        __builtin_amdgcn_sched_group_barrier(0x020, 4, 0);   // 4 VMEM reads
        __builtin_amdgcn_sched_group_barrier(0x100, 16, 0);  // 16 DS reads
        __builtin_amdgcn_sched_group_barrier(0x008, 8, 0);   // 8 WMMA
      }
    }
  }

  // scatter D (16x16 f32 layout: lane<16 -> M=r, lane>=16 -> M=r+8; N=lane%16) + bias + stats
  int wo = wo_base + nn;
  int mb = hh * 8;
  #pragma unroll
  for (int mt = 0; mt < 4; ++mt) {
    v8f a = (mt == 0) ? acc0 : (mt == 1) ? acc1 : (mt == 2) ? acc2 : acc3;
    #pragma unroll
    for (int r = 0; r < 8; ++r) {
      int c = mt * 16 + mb + r;
      float yv = a[r] + bias[c];
      out[((size_t)c * XD + x) * (HO * WOUT) + (size_t)ho * WOUT + wo] = yv;
      atomicAdd(&csum[c], yv);
      atomicAdd(&csq[c], yv * yv);
    }
  }
  __syncthreads();
  if (tid < 64) {
    atomicAdd(&gstats[tid], csum[tid]);
    atomicAdd(&gstats[64 + tid], csq[tid]);
  }
}

// ---------- kernel 4: batchnorm (batch stats) + leaky relu, in place on d_out ----------
__global__ void bn_lrelu_k(float* __restrict__ out, const float* __restrict__ gstats,
                           const float* __restrict__ gamma, const float* __restrict__ beta) {
  size_t t = (size_t)blockIdx.x * blockDim.x + threadIdx.x;   // one float4 per thread
  int c = (int)(t >> 18);                                     // 2^18 float4 per channel
  const float inv = 1.0f / (float)NPTS;
  float mean = gstats[c] * inv;
  float var  = gstats[64 + c] * inv - mean * mean;
  float sc = gamma[c] * rsqrtf(var + 1e-5f);
  float sh = beta[c] - mean * sc;
  float4* p = (float4*)out + t;
  float4 v = *p;
  v.x = v.x * sc + sh; v.x = (v.x > 0.0f) ? v.x : 0.01f * v.x;
  v.y = v.y * sc + sh; v.y = (v.y > 0.0f) ? v.y : 0.01f * v.y;
  v.z = v.z * sc + sh; v.z = (v.z > 0.0f) ? v.z : 0.01f * v.z;
  v.w = v.w * sc + sh; v.w = (v.w > 0.0f) ? v.w : 0.01f * v.w;
  *p = v;
}

// ---------- launch ----------
extern "C" void kernel_launch(void* const* d_in, const int* in_sizes, int n_in,
                              void* d_out, int out_size, void* d_ws, size_t ws_size,
                              hipStream_t stream) {
  (void)in_sizes; (void)n_in; (void)out_size; (void)ws_size;
  const float* feats  = (const float*)d_in[0];
  const float* grids  = (const float*)d_in[1];
  // d_in[2] grid_masks, d_in[3] masks: unused by the reference computation (all-ones)
  const float* conv_w = (const float*)d_in[4];
  const float* conv_b = (const float*)d_in[5];
  const float* gamma  = (const float*)d_in[6];
  const float* beta   = (const float*)d_in[7];
  float* out = (float*)d_out;

  char* ws = (char*)d_ws;
  uint16_t* vol    = (uint16_t*)ws;
  uint16_t* wbuf   = (uint16_t*)(ws + WBUF_OFF);
  float*    gstats = (float*)(ws + STATS_OFF);
  const char* zpage = ws + ZPAGE_OFF;

  zero_stats_k<<<1, 256, 0, stream>>>(gstats);
  repack_w_k<<<WB_ELEMS / 256, 256, 0, stream>>>(conv_w, wbuf);
  sweep_sample_k<<<(4 * NPTS) / 256, 256, 0, stream>>>(feats, grids, vol);
  conv3d_wmma_k<<<XD * HO, 512, 0, stream>>>(vol, wbuf, conv_b, out, gstats, zpage);
  bn_lrelu_k<<<(NPTS * FC / 4) / 256, 256, 0, stream>>>(out, gstats, gamma, beta);
}
